// RelativePositionalEncoding_79061757985077
// MI455X (gfx1250) — matrix operand order repaired and checked by reading, not compiled, not run on hardware
//
#include <hip/hip_runtime.h>

typedef __attribute__((ext_vector_type(16))) _Float16 v16h;
typedef __attribute__((ext_vector_type(8)))  _Float16 v8h;
typedef __attribute__((ext_vector_type(8)))  float    v8f;

#define NPTS  1024
#define HID   64
#define NH    8
#define BATCH 2
#define KI    16   // i-values processed per wave (amortizes A-row + W2 fragment loads)

// ---------------------------------------------------------------------------
// Kernel 1: U[b,n,h] = pts[b,n]·W1[:,h]  (f16), V = U - b1 (f16)
// Removes the 3->64 matmul from the O(N^2) loop: rel·W1 = u[j] - u[i].
// ---------------------------------------------------------------------------
__global__ __launch_bounds__(256) void prep_uv(
    const float* __restrict__ xyz, const float* __restrict__ W1,
    const float* __restrict__ b1,
    _Float16* __restrict__ U, _Float16* __restrict__ V)
{
  int idx = blockIdx.x * blockDim.x + threadIdx.x;   // [0, B*N*HID) = [0, 131072)
  int h = idx & (HID - 1);
  int n = (idx >> 6) & (NPTS - 1);
  int b = idx >> 16;                                 // N*HID = 65536
  float x0 = xyz[b * 3 * NPTS + 0 * NPTS + n];
  float x1 = xyz[b * 3 * NPTS + 1 * NPTS + n];
  float x2 = xyz[b * 3 * NPTS + 2 * NPTS + n];
  float u = x0 * W1[h] + x1 * W1[HID + h] + x2 * W1[2 * HID + h];
  U[idx] = (_Float16)u;
  V[idx] = (_Float16)(u - b1[h]);
}

// ---------------------------------------------------------------------------
// Kernel 2: per wave, one (b, 16-j tile); loop over KI i's.
// out[b,k,i,jtile] tile = relu(U[jtile]-V[i]) (16x64 f16, A) x W2pad (64x16, B)
// via two V_WMMA_F32_16X16X32_F16 (K split 0..31 / 32..63).
// ---------------------------------------------------------------------------
__global__ __launch_bounds__(128) void pair_bias(
    const _Float16* __restrict__ U, const _Float16* __restrict__ V,
    const float* __restrict__ W2, const float* __restrict__ b2f,
    float* __restrict__ out)
{
  const int lane = threadIdx.x & 31;
  const int wave = threadIdx.x >> 5;
  int task = blockIdx.x * 4 + wave;                 // 8192 wave tasks total

  const int IG = NPTS / KI;                         // 64 i-groups
  const int JT = NPTS / 16;                         // 64 j-tiles
  int ig = task % IG; task /= IG;
  int jt = task % JT;
  int b  = task / JT;

  const int n   = lane & 15;                        // D col = head index (0..7 valid)
  const int hi  = lane >> 4;                        // lane half
  const int kb  = hi * 8;                           // A-layout feature base (K subset)
  const int kb2 = hi * 16;                          // unused (B fragments are built from kb)

  // ---- B fragments: W2 (64x8) padded to 64x16, per ISA B layout.
  //      element e of v16h <-> K = kb + e (+8 if e >= 8) (+32 for chunk 1), N = n.
  v16h bm0, bm1;
#pragma unroll
  for (int e = 0; e < 16; ++e) {
    float w0 = (n < NH) ? W2[(kb + e + (e >= 8 ? 8 : 0)) * NH + n]      : 0.0f;
    float w1 = (n < NH) ? W2[(kb + e + (e >= 8 ? 8 : 0) + 32) * NH + n] : 0.0f;
    bm0[e] = (_Float16)w0;
    bm1[e] = (_Float16)w1;
  }
  const float bias = (n < NH) ? b2f[n] : 0.0f;

  // ---- A-side u row for this lane (row M = j), loaded once per wave.
  //      A layout: elems 0..7 -> K = kb+e, elems 8..15 -> K = 16+kb+e (+32 chunk1)
  const int jbase = jt * 16;
  const _Float16* urow = U + ((size_t)(b * NPTS + jbase + n)) * HID;
  v8h u0 = *(const v8h*)(urow + kb);
  v8h u1 = *(const v8h*)(urow + kb + 16);
  v8h u2 = *(const v8h*)(urow + kb + 32);
  v8h u3 = *(const v8h*)(urow + kb + 48);

  const int ibase = ig * KI;
  for (int ii = 0; ii < KI; ++ii) {
    const int i = ibase + ii;
    const _Float16* vrow = V + ((size_t)(b * NPTS + i)) * HID;
    v8h v0 = *(const v8h*)(vrow + kb);
    v8h v1 = *(const v8h*)(vrow + kb + 16);
    v8h v2 = *(const v8h*)(vrow + kb + 32);
    v8h v3 = *(const v8h*)(vrow + kb + 48);

    // h = relu(u_j - v_i), assembled directly in the WMMA A fragment layout
    v16h a0, a1;
#pragma unroll
    for (int e = 0; e < 8; ++e) {
      _Float16 t0 = u0[e] - v0[e];
      _Float16 t1 = u1[e] - v1[e];
      _Float16 t2 = u2[e] - v2[e];
      _Float16 t3 = u3[e] - v3[e];
      a0[e]     = t0 > (_Float16)0 ? t0 : (_Float16)0;
      a0[e + 8] = t1 > (_Float16)0 ? t1 : (_Float16)0;
      a1[e]     = t2 > (_Float16)0 ? t2 : (_Float16)0;
      a1[e + 8] = t3 > (_Float16)0 ? t3 : (_Float16)0;
    }

    // D(16 j x 16 k) += A(16x32) x B(32x16), K = 0..31 then 32..63
    v8f acc = {};
    acc = __builtin_amdgcn_wmma_f32_16x16x32_f16(false, a0, false, bm0,
                                                 (short)0, acc, false, false);
    acc = __builtin_amdgcn_wmma_f32_16x16x32_f16(false, a1, false, bm1,
                                                 (short)0, acc, false, false);

    // D: VGPR r -> j = jbase + r + 8*hi ; N = n = head. Store 8 contiguous j's.
    if (n < NH) {
      float* dst = out + (((size_t)(b * NH + n) * NPTS) + i) * NPTS + jbase + hi * 8;
      float4 s0 = make_float4(acc[0] + bias, acc[1] + bias,
                              acc[2] + bias, acc[3] + bias);
      float4 s1 = make_float4(acc[4] + bias, acc[5] + bias,
                              acc[6] + bias, acc[7] + bias);
      *(float4*)(dst)     = s0;
      *(float4*)(dst + 4) = s1;
    }
  }
}

extern "C" void kernel_launch(void* const* d_in, const int* in_sizes, int n_in,
                              void* d_out, int out_size, void* d_ws, size_t ws_size,
                              hipStream_t stream) {
  const float* xyz = (const float*)d_in[0];
  const float* W1  = (const float*)d_in[1];
  const float* b1  = (const float*)d_in[2];
  const float* W2  = (const float*)d_in[3];
  const float* b2  = (const float*)d_in[4];
  float* out = (float*)d_out;

  _Float16* U = (_Float16*)d_ws;                      // 2*1024*64 halfs = 256 KB
  _Float16* V = U + (size_t)BATCH * NPTS * HID;       // another 256 KB

  prep_uv<<<(BATCH * NPTS * HID) / 256, 256, 0, stream>>>(xyz, W1, b1, U, V);

  const int tasks = BATCH * (NPTS / 16) * (NPTS / KI);  // 8192 wave tasks
  pair_bias<<<tasks / 4, 128, 0, stream>>>(U, V, W2, b2, out);
}
